// BaselineModel_64536178590371
// MI455X (gfx1250) — matrix operand order repaired and checked by reading, not compiled, not run on hardware
//
#include <hip/hip_runtime.h>

// ---------------------------------------------------------------------------
// Types & helpers
// ---------------------------------------------------------------------------
typedef __attribute__((ext_vector_type(16))) __bf16 v16bf;
typedef __attribute__((ext_vector_type(8)))  float  v8f;
typedef __attribute__((ext_vector_type(4)))  int    v4i;
typedef __attribute__((ext_vector_type(8)))  int    v8i;

__device__ __forceinline__ unsigned short f2bf(float f) {
    unsigned int u = __builtin_bit_cast(unsigned int, f);
    unsigned int r = u + 0x7FFFu + ((u >> 16) & 1u);   // round-to-nearest-even
    return (unsigned short)(r >> 16);
}
__device__ __forceinline__ __bf16 us2bf(unsigned short s) {
    return __builtin_bit_cast(__bf16, s);
}
__device__ __forceinline__ v8f v8f_zero() {
    v8f z = {0.f,0.f,0.f,0.f,0.f,0.f,0.f,0.f};
    return z;
}

// async copy of 16 bytes global -> LDS (ASYNCcnt-tracked).
// lds_base is passed so the LDS object ESCAPES into the asm: without it LLVM
// proves the shared array is never stored and folds all ds loads to undef.
__device__ __forceinline__ void async_b128(unsigned int lds_byte_off, const void* gsrc,
                                           void* lds_base) {
    unsigned long long ga = (unsigned long long)gsrc;
    unsigned long long lb = (unsigned long long)lds_base;
    asm volatile("global_load_async_to_lds_b128 %0, %1, off"
                 :: "v"(lds_byte_off), "v"(ga), "v"(lb) : "memory");
}
__device__ __forceinline__ void wait_async0() {
    asm volatile("s_wait_asynccnt 0x0" ::: "memory");
}

// Tensor Data Mover: 2D tile (dim0 elements x rows) of bf16 from global to
// LDS, with LDS padding of 4 DWORDs (8 elems) after every 16 DWORDs (32
// elems) -> matches a padded LDS row stride of 40 elements for dim0==32.
// Issued once per wave (EXEC-independent); caller gates to one wave.
__device__ __forceinline__ void tdm_load_2d_pad(
    unsigned int lds_byte_addr, const void* gsrc,
    unsigned int dim0, unsigned int rows,
    unsigned int tensor_d0, unsigned int tensor_d1,
    unsigned int stride_elems, void* lds_base)
{
    unsigned long long ga = (unsigned long long)gsrc;
    v4i g0;
    g0[0] = 1;                                              // count=1, user D#
    g0[1] = (int)lds_byte_addr;                             // lds_addr
    g0[2] = (int)(unsigned int)(ga & 0xffffffffu);          // global_addr lo
    g0[3] = (int)((unsigned int)((ga >> 32) & 0x01ffffffu)  // global_addr hi
                  | (2u << 30));                            // type=2 (image)
    v8i g1;
    g1[0] = (int)((1u << 16)      // data_size = 2 bytes
                | (1u << 20)      // pad_enable
                | (3u << 22)      // pad_interval: 16 DWORDs
                | (3u << 25));    // pad_amount:   4 DWORDs
    g1[1] = (int)((tensor_d0 & 0xffffu) << 16);             // tensor_dim0 lo16
    g1[2] = (int)((tensor_d0 >> 16) | ((tensor_d1 & 0xffffu) << 16));
    g1[3] = (int)((tensor_d1 >> 16) | (dim0 << 16));        // tile_dim0
    g1[4] = (int)rows;                                      // tile_dim1 (dim2=0)
    g1[5] = (int)stride_elems;                              // tensor_dim0_stride lo
    g1[6] = 0;
    g1[7] = 0;
    unsigned long long lb = (unsigned long long)lds_base;
    asm volatile("tensor_load_to_lds %0, %1"
                 :: "s"(g0), "s"(g1), "v"(lb) : "memory");
}

#define B_  4
#define S_  192
#define H_  768
#define L_  12
#define NH_ 12
#define DH_ 64
#define F_  3072
#define MID_ 1024
#define TOK_ (B_*S_)   // 768

// ---------------------------------------------------------------------------
// Tiled fp32 -> bf16 TRANSPOSING convert:  out[N,K] = bf16(in[K,N])
// ---------------------------------------------------------------------------
__global__ __launch_bounds__(256) void cvt_t_kernel(
    const float* __restrict__ in, unsigned short* __restrict__ out, int K, int N)
{
    __shared__ unsigned short tile[32][33];     // [k-local][n-local], padded
    const int kt = blockIdx.y * 32, nt = blockIdx.x * 32;
    const int t  = threadIdx.x;
    const int tr = t >> 3;          // 0..31
    const int tc = (t & 7) * 4;     // 0,4,...,28
    #pragma unroll
    for (int i = 0; i < 4; ++i)
        tile[tr][tc + i] = f2bf(in[(size_t)(kt + tr) * N + nt + tc + i]);
    __syncthreads();
    #pragma unroll
    for (int i = 0; i < 4; ++i)
        out[(size_t)(nt + tr) * K + kt + tc + i] = tile[tc + i][tr];
}

// ---------------------------------------------------------------------------
// Generic bf16 WMMA GEMM: C[M,N] = act(A[M,K] @ Wt[N,K]^T + bias)
// Block tile 128x64, 8 waves. Tiles staged by TWO Tensor-DMA descriptors
// (wave 0 issues, s_wait_tensorcnt, block barrier). act: 0=none, 1=GELU.
// ---------------------------------------------------------------------------
#define BM 128
#define BN 64
#define BK 32
#define ASTR (BK + 8)               // 40-elem padded LDS stride (16B-aligned)
#define BSTR (BK + 8)
#define BBASE (BM * ASTR)           // 5120 elements -> byte 10240

__global__ __launch_bounds__(256) void gemm_bf16_kernel(
    const unsigned short* __restrict__ A,   // [M,K] bf16
    const unsigned short* __restrict__ Wt,  // [N,K] bf16 (pre-transposed)
    const float* __restrict__ bias,         // [N]
    float* __restrict__ outF,               // [M,N] or null
    unsigned short* __restrict__ outB,      // [M,N] bf16 or null
    int M, int N, int K, int act)
{
    __shared__ unsigned short smem[BM*ASTR + BN*BSTR];
    const int tid  = threadIdx.x;
    const int wave = tid >> 5;
    const int lane = tid & 31;
    const int m0   = blockIdx.y * BM + wave * 16;
    const int n0   = blockIdx.x * BN;

    v8f acc[4] = { v8f_zero(), v8f_zero(), v8f_zero(), v8f_zero() };

    const int arow = (lane & 15);
    const int akb  = (lane < 16) ? 0 : 8;    // A frag K-half base
    const int bcol = (lane & 15);
    const int bkb  = (lane < 16) ? 0 : 8;    // B frag K-half base

    for (int k0 = 0; k0 < K; k0 += BK) {
        if (wave == 0) {
            // A tile: 128 rows x 32 elems, row stride K
            tdm_load_2d_pad(0u,
                            A + (size_t)(blockIdx.y*BM) * K + k0,
                            BK, BM, (unsigned)K, (unsigned)M, (unsigned)K,
                            (void*)smem);
            // Wt tile: 64 rows x 32 elems, row stride K
            tdm_load_2d_pad((unsigned)(BBASE * 2),
                            Wt + (size_t)n0 * K + k0,
                            BK, BN, (unsigned)K, (unsigned)N, (unsigned)K,
                            (void*)smem);
            __builtin_amdgcn_s_wait_tensorcnt(0);
        }
        __syncthreads();

        v16bf af;
        #pragma unroll
        for (int i = 0; i < 16; ++i) {
            int kk = ((i < 8) ? i : i + 8) + akb;
            af[i] = us2bf(smem[(wave*16 + arow)*ASTR + kk]);
        }
        #pragma unroll
        for (int j = 0; j < 4; ++j) {
            v16bf bf;
            #pragma unroll
            for (int i = 0; i < 16; ++i) {
                bf[i] = us2bf(smem[BBASE + (j*16 + bcol)*BSTR + ((i < 8) ? i : i + 8) + bkb]);
            }
            acc[j] = __builtin_amdgcn_wmma_f32_16x16x32_bf16(
                false, af, false, bf, (short)0, acc[j], false, false);
        }
        __syncthreads();
    }

    const int radd = (lane < 16) ? 0 : 8;
    #pragma unroll
    for (int j = 0; j < 4; ++j) {
        int col  = n0 + j*16 + (lane & 15);
        float bv = bias ? bias[col] : 0.0f;
        #pragma unroll
        for (int r = 0; r < 8; ++r) {
            int row = m0 + r + radd;
            float v = acc[j][r] + bv;
            if (act == 1) v = 0.5f * v * (1.0f + erff(v * 0.70710678118f));
            size_t idx = (size_t)row * N + col;
            if (outF) outF[idx] = v;
            if (outB) outB[idx] = f2bf(v);
        }
    }
}

// ---------------------------------------------------------------------------
// Attention scores: att[bh,q,k] = (Q[q,:]·K[k,:]) * scale + abias[k]
// One block per (b,h). Q,K async-staged to LDS (ASYNCcnt path).
// ---------------------------------------------------------------------------
#define QSTR (DH_ + 8)              // 72 elements, rows 144B (16B-aligned)
#define KSBASE (S_ * QSTR)          // 13824 elements

__global__ __launch_bounds__(256) void attn_scores_kernel(
    const unsigned short* __restrict__ qb,  // [TOK,H] bf16
    const unsigned short* __restrict__ kb,  // [TOK,H] bf16
    const int* __restrict__ amask,          // [B,S]
    float* __restrict__ att)                // [B*NH,S,S]
{
    __shared__ unsigned short smem[2*S_*QSTR + 2*S_];   // Qs | Ks | abias(float)
    float* abias = (float*)(smem + 2*S_*QSTR);
    const int bh = blockIdx.x;
    const int b = bh / NH_, h = bh % NH_;
    const int tid = threadIdx.x, wave = tid >> 5, lane = tid & 31;

    for (int c = tid; c < (S_*DH_)/8; c += 256) {
        int s = c >> 3;
        int d = (c & 7) * 8;
        size_t g = (size_t)(b*S_ + s) * H_ + h*DH_ + d;
        async_b128((unsigned int)((s*QSTR + d) * 2),          qb + g, (void*)smem);
        async_b128((unsigned int)((KSBASE + s*QSTR + d) * 2), kb + g, (void*)smem);
    }
    for (int s = tid; s < S_; s += 256)
        abias[s] = (1.0f - (float)amask[b*S_ + s]) * -1e4f;
    wait_async0();
    __syncthreads();

    const float scale = 0.125f;     // 1/sqrt(64)
    const int akb = (lane < 16) ? 0 : 8;
    const int bkb = (lane < 16) ? 0 : 8;

    for (int t = wave; t < (S_/16)*(S_/16); t += 8) {   // 144 tiles
        int mt = t / (S_/16), nt = t % (S_/16);
        v8f acc = v8f_zero();
        int qrow = mt*16 + (lane & 15);
        int krow = nt*16 + (lane & 15);
        #pragma unroll
        for (int kc = 0; kc < DH_/32; ++kc) {
            v16bf af, bf;
            #pragma unroll
            for (int i = 0; i < 16; ++i) {
                int ka = kc*32 + ((i < 8) ? i : i + 8) + akb;
                af[i] = us2bf(smem[qrow*QSTR + ka]);
                int kv = kc*32 + ((i < 8) ? i : i + 8) + bkb;
                bf[i] = us2bf(smem[KSBASE + krow*QSTR + kv]);
            }
            acc = __builtin_amdgcn_wmma_f32_16x16x32_bf16(
                false, af, false, bf, (short)0, acc, false, false);
        }
        int radd = (lane < 16) ? 0 : 8;
        int col  = nt*16 + (lane & 15);
        #pragma unroll
        for (int r = 0; r < 8; ++r) {
            int row = mt*16 + r + radd;
            att[((size_t)bh*S_ + row)*S_ + col] = acc[r]*scale + abias[col];
        }
    }
}

// ---------------------------------------------------------------------------
// Row softmax over 192 columns, emits bf16 probabilities
// ---------------------------------------------------------------------------
__global__ __launch_bounds__(192) void softmax_kernel(
    const float* __restrict__ att, unsigned short* __restrict__ attb)
{
    __shared__ float red[256];
    const int row = blockIdx.x;
    const int t = threadIdx.x;
    float v = att[(size_t)row*S_ + t];
    red[t] = v; if (t < 64) red[192 + t] = -1e30f;
    __syncthreads();
    for (int s = 128; s > 0; s >>= 1) { if (t < s) red[t] = fmaxf(red[t], red[t+s]); __syncthreads(); }
    float mx = red[0]; __syncthreads();
    float e = __expf(v - mx);
    red[t] = e; if (t < 64) red[192 + t] = 0.0f;
    __syncthreads();
    for (int s = 128; s > 0; s >>= 1) { if (t < s) red[t] += red[t+s]; __syncthreads(); }
    attb[(size_t)row*S_ + t] = f2bf(e / red[0]);
}

// ---------------------------------------------------------------------------
// ctx = att @ V per head; V staged TRANSPOSED in LDS so B-fragments are
// contiguous (2x ds_load_b128). A-fragments load straight from global bf16.
// ---------------------------------------------------------------------------
#define VSTR (S_ + 8)               // 200 elements, rows 400B (16B-aligned)

__global__ __launch_bounds__(256) void attn_av_kernel(
    const unsigned short* __restrict__ attb,  // [B*NH,S,S] bf16
    const unsigned short* __restrict__ vb,    // [TOK,H]   bf16
    unsigned short* __restrict__ ctxb)        // [TOK,H]   bf16
{
    __shared__ unsigned short Vst[DH_ * VSTR];  // [d][s]
    const int bh = blockIdx.x;
    const int b = bh / NH_, h = bh % NH_;
    const int tid = threadIdx.x, wave = tid >> 5, lane = tid & 31;

    for (int c = tid; c < S_*DH_; c += 256) {
        int s = c >> 6;             // /64
        int d = c & 63;
        Vst[d*VSTR + s] = vb[(size_t)(b*S_ + s)*H_ + h*DH_ + d];
    }
    __syncthreads();

    const int akb = (lane < 16) ? 0 : 8;
    const int bkb = (lane < 16) ? 0 : 8;
    for (int t = wave; t < (S_/16)*(DH_/16); t += 8) {    // 48 tiles
        int mt = t / (DH_/16), nt = t % (DH_/16);
        v8f acc = v8f_zero();
        int arow = mt*16 + (lane & 15);
        int bn   = nt*16 + (lane & 15);
        const unsigned short* ap = attb + ((size_t)bh*S_ + arow)*S_;
        #pragma unroll
        for (int kc = 0; kc < S_/32; ++kc) {              // 6 K-chunks
            v16bf af, bf;
            #pragma unroll
            for (int i = 0; i < 16; ++i) {
                int ka = kc*32 + ((i < 8) ? i : i + 8) + akb;
                af[i] = us2bf(ap[ka]);
                int kv = kc*32 + ((i < 8) ? i : i + 8) + bkb;
                bf[i] = us2bf(Vst[bn*VSTR + kv]);
            }
            acc = __builtin_amdgcn_wmma_f32_16x16x32_bf16(
                false, af, false, bf, (short)0, acc, false, false);
        }
        int radd = (lane < 16) ? 0 : 8;
        int col  = nt*16 + (lane & 15);
        #pragma unroll
        for (int r = 0; r < 8; ++r) {
            int row = mt*16 + r + radd;
            ctxb[(size_t)(b*S_ + row)*H_ + h*DH_ + col] = f2bf(acc[r]);
        }
    }
}

// ---------------------------------------------------------------------------
// LayerNorm over H=768: x_out = LN(x_in + res) * s + b ; also bf16 copy
// ---------------------------------------------------------------------------
__global__ __launch_bounds__(256) void ln_kernel(
    const float* __restrict__ xin, const float* __restrict__ res,
    const float* __restrict__ s, const float* __restrict__ bpar,
    float* __restrict__ xoutF, unsigned short* __restrict__ xoutB)
{
    __shared__ float red[256];
    const int row = blockIdx.x, t = threadIdx.x;
    float v[3];
    float lsum = 0.0f;
    #pragma unroll
    for (int i = 0; i < 3; ++i) {
        int col = t + i*256;
        float val = xin[(size_t)row*H_ + col];
        if (res) val += res[(size_t)row*H_ + col];
        v[i] = val; lsum += val;
    }
    red[t] = lsum; __syncthreads();
    for (int st = 128; st > 0; st >>= 1) { if (t < st) red[t] += red[t+st]; __syncthreads(); }
    float mu = red[0] * (1.0f / H_); __syncthreads();
    float lvar = 0.0f;
    #pragma unroll
    for (int i = 0; i < 3; ++i) { float d = v[i] - mu; lvar += d*d; }
    red[t] = lvar; __syncthreads();
    for (int st = 128; st > 0; st >>= 1) { if (t < st) red[t] += red[t+st]; __syncthreads(); }
    float inv = rsqrtf(red[0] * (1.0f / H_) + 1e-12f);
    #pragma unroll
    for (int i = 0; i < 3; ++i) {
        int col = t + i*256;
        float o = (v[i] - mu) * inv * s[col] + bpar[col];
        xoutF[(size_t)row*H_ + col] = o;
        if (xoutB) xoutB[(size_t)row*H_ + col] = f2bf(o);
    }
}

// ---------------------------------------------------------------------------
// Embedding gather/add
// ---------------------------------------------------------------------------
__global__ __launch_bounds__(256) void emb_kernel(
    const int* __restrict__ ids, const int* __restrict__ segs,
    const float* __restrict__ we, const float* __restrict__ pe,
    const float* __restrict__ te, float* __restrict__ x)
{
    const int row = blockIdx.x, t = threadIdx.x;
    const int sp = row % S_;
    const int id = ids[row], sg = segs[row];
    #pragma unroll
    for (int i = 0; i < 3; ++i) {
        int col = t + i*256;
        x[(size_t)row*H_ + col] = we[(size_t)id*H_ + col]
                                + pe[(size_t)sp*H_ + col]
                                + te[(size_t)sg*H_ + col];
    }
}

// ---------------------------------------------------------------------------
// Pairwise tanh scorer + masked BCE accumulation.
// ---------------------------------------------------------------------------
__global__ __launch_bounds__(192) void pair_kernel(
    const float* __restrict__ ha, const float* __restrict__ ht,
    const float* __restrict__ ow, const float* __restrict__ ob,
    const float* __restrict__ target, const unsigned char* __restrict__ maskp,
    float* __restrict__ outp, float* __restrict__ accum)
{
    __shared__ float has[MID_];
    __shared__ float ws[MID_];
    __shared__ float red[256];
    const int bq = blockIdx.x;
    const int b = bq / S_;
    const int t = threadIdx.x;
    for (int i = t; i < MID_; i += 192) {
        has[i] = ha[(size_t)bq*MID_ + i];
        ws[i]  = ow[i];
    }
    __syncthreads();
    const float* htr = ht + (size_t)(b*S_ + t) * MID_;
    float acc = 0.0f;
    for (int m = 0; m < MID_; ++m)
        acc += tanhf(has[m] + htr[m]) * ws[m];
    float logit = acc + ob[0];
    size_t oidx = (size_t)bq*S_ + t;
    outp[oidx] = logit;
    float mk = (float)maskp[oidx];
    float tg = target[oidx];
    float bce = fmaxf(logit, 0.0f) - logit*tg + log1pf(__expf(-fabsf(logit)));

    red[t] = bce * mk; if (t < 64) red[192 + t] = 0.0f;
    __syncthreads();
    for (int s = 128; s > 0; s >>= 1) { if (t < s) red[t] += red[t+s]; __syncthreads(); }
    if (t == 0) atomicAdd(&accum[0], red[0]);
    __syncthreads();
    red[t] = mk; if (t < 64) red[192 + t] = 0.0f;
    __syncthreads();
    for (int s = 128; s > 0; s >>= 1) { if (t < s) red[t] += red[t+s]; __syncthreads(); }
    if (t == 0) atomicAdd(&accum[1], red[0]);
}

__global__ void zero2_kernel(float* a) { if (threadIdx.x < 2) a[threadIdx.x] = 0.0f; }
__global__ void loss_kernel(const float* __restrict__ accum, float* __restrict__ loss) {
    loss[0] = accum[0] / fmaxf(accum[1], 1.0f);
}

// ---------------------------------------------------------------------------
// Host orchestration
// ---------------------------------------------------------------------------
extern "C" void kernel_launch(void* const* d_in, const int* in_sizes, int n_in,
                              void* d_out, int out_size, void* d_ws, size_t ws_size,
                              hipStream_t stream)
{
    (void)in_sizes; (void)n_in; (void)out_size; (void)ws_size;
    const int*   input_ids = (const int*)  d_in[0];
    const int*   amask     = (const int*)  d_in[1];
    const int*   seg_ids   = (const int*)  d_in[2];
    const unsigned char* maskp = (const unsigned char*)d_in[3];
    const float* target    = (const float*)d_in[4];
    const float* word_emb  = (const float*)d_in[5];
    const float* pos_emb   = (const float*)d_in[6];
    const float* type_emb  = (const float*)d_in[7];
    const float* emb_ln_s  = (const float*)d_in[8];
    const float* emb_ln_b  = (const float*)d_in[9];
    const float* Wq = (const float*)d_in[10]; const float* bq = (const float*)d_in[11];
    const float* Wk = (const float*)d_in[12]; const float* bk = (const float*)d_in[13];
    const float* Wv = (const float*)d_in[14]; const float* bv = (const float*)d_in[15];
    const float* Wo = (const float*)d_in[16]; const float* bo = (const float*)d_in[17];
    const float* ln1_s = (const float*)d_in[18]; const float* ln1_b = (const float*)d_in[19];
    const float* W1 = (const float*)d_in[20]; const float* b1 = (const float*)d_in[21];
    const float* W2 = (const float*)d_in[22]; const float* b2 = (const float*)d_in[23];
    const float* ln2_s = (const float*)d_in[24]; const float* ln2_b = (const float*)d_in[25];
    const float* ana_w = (const float*)d_in[26]; const float* ana_b = (const float*)d_in[27];
    const float* ant_w = (const float*)d_in[28]; const float* ant_b = (const float*)d_in[29];
    const float* out_w = (const float*)d_in[30]; const float* out_b = (const float*)d_in[31];

    float* loss   = (float*)d_out;
    float* output = (float*)d_out + 1;

    // ---- carve workspace (256B aligned) ----
    char* p = (char*)d_ws;
    auto carve = [&p](size_t bytes) -> void* {
        void* r = (void*)p;
        p += (bytes + 255) & ~(size_t)255;
        return r;
    };
    float*          x    = (float*)         carve((size_t)TOK_*H_*4);
    unsigned short* xb   = (unsigned short*)carve((size_t)TOK_*H_*2);
    unsigned short* wb   = (unsigned short*)carve((size_t)H_*F_*2);     // weight staging (transposed)
    unsigned short* qbuf = (unsigned short*)carve((size_t)TOK_*H_*2);
    unsigned short* kbuf = (unsigned short*)carve((size_t)TOK_*H_*2);
    unsigned short* vbuf = (unsigned short*)carve((size_t)TOK_*H_*2);
    unsigned short* ctxb = (unsigned short*)carve((size_t)TOK_*H_*2);
    float*          tmp  = (float*)         carve((size_t)TOK_*H_*4);
    unsigned short* ffb  = (unsigned short*)carve((size_t)TOK_*F_*2);
    float*          att  = (float*)         carve((size_t)B_*NH_*S_*S_*4);
    unsigned short* attb = (unsigned short*)carve((size_t)B_*NH_*S_*S_*2);
    float*          ha   = (float*)         carve((size_t)TOK_*MID_*4);
    float*          ht   = (float*)         carve((size_t)TOK_*MID_*4);
    float*          acc2 = (float*)         carve(2*4);

    const size_t wHH = (size_t)H_*H_;
    const size_t wHF = (size_t)H_*F_;
    dim3 gHH(H_/BN,  TOK_/BM);    // N=768
    dim3 gHF(F_/BN,  TOK_/BM);    // N=3072
    dim3 gHM(MID_/BN, TOK_/BM);   // N=1024
    dim3 tHH(H_/32,  H_/32);      // transpose grids: (N/32, K/32)
    dim3 tHF(F_/32,  H_/32);      // W1: K=H, N=F
    dim3 tFH(H_/32,  F_/32);      // W2: K=F, N=H
    dim3 tHM(MID_/32, H_/32);     // ana/ant: K=H, N=MID

    zero2_kernel<<<1, 32, 0, stream>>>(acc2);

    // embeddings + LN
    emb_kernel<<<TOK_, 256, 0, stream>>>(input_ids, seg_ids, word_emb, pos_emb, type_emb, x);
    ln_kernel<<<TOK_, 256, 0, stream>>>(x, nullptr, emb_ln_s, emb_ln_b, x, xb);

    for (int l = 0; l < L_; ++l) {
        // Q, K, V projections (bf16 outputs only)
        cvt_t_kernel<<<tHH, 256, 0, stream>>>(Wq + l*wHH, wb, H_, H_);
        gemm_bf16_kernel<<<gHH, 256, 0, stream>>>(xb, wb, bq + l*H_, nullptr, qbuf, TOK_, H_, H_, 0);
        cvt_t_kernel<<<tHH, 256, 0, stream>>>(Wk + l*wHH, wb, H_, H_);
        gemm_bf16_kernel<<<gHH, 256, 0, stream>>>(xb, wb, bk + l*H_, nullptr, kbuf, TOK_, H_, H_, 0);
        cvt_t_kernel<<<tHH, 256, 0, stream>>>(Wv + l*wHH, wb, H_, H_);
        gemm_bf16_kernel<<<gHH, 256, 0, stream>>>(xb, wb, bv + l*H_, nullptr, vbuf, TOK_, H_, H_, 0);

        // attention
        attn_scores_kernel<<<B_*NH_, 256, 0, stream>>>(qbuf, kbuf, amask, att);
        softmax_kernel<<<B_*NH_*S_, 192, 0, stream>>>(att, attb);
        attn_av_kernel<<<B_*NH_, 256, 0, stream>>>(attb, vbuf, ctxb);

        // output projection + residual LN
        cvt_t_kernel<<<tHH, 256, 0, stream>>>(Wo + l*wHH, wb, H_, H_);
        gemm_bf16_kernel<<<gHH, 256, 0, stream>>>(ctxb, wb, bo + l*H_, tmp, nullptr, TOK_, H_, H_, 0);
        ln_kernel<<<TOK_, 256, 0, stream>>>(x, tmp, ln1_s + l*H_, ln1_b + l*H_, x, xb);

        // FFN
        cvt_t_kernel<<<tHF, 256, 0, stream>>>(W1 + l*wHF, wb, H_, F_);
        gemm_bf16_kernel<<<gHF, 256, 0, stream>>>(xb, wb, b1 + l*F_, nullptr, ffb, TOK_, F_, H_, 1);
        cvt_t_kernel<<<tFH, 256, 0, stream>>>(W2 + l*wHF, wb, F_, H_);
        gemm_bf16_kernel<<<gHH, 256, 0, stream>>>(ffb, wb, b2 + l*H_, tmp, nullptr, TOK_, H_, F_, 0);
        ln_kernel<<<TOK_, 256, 0, stream>>>(x, tmp, ln2_s + l*H_, ln2_b + l*H_, x, xb);
    }

    // pairwise scorer heads
    cvt_t_kernel<<<tHM, 256, 0, stream>>>(ana_w, wb, H_, MID_);
    gemm_bf16_kernel<<<gHM, 256, 0, stream>>>(xb, wb, ana_b, ha, nullptr, TOK_, MID_, H_, 0);
    cvt_t_kernel<<<tHM, 256, 0, stream>>>(ant_w, wb, H_, MID_);
    gemm_bf16_kernel<<<gHM, 256, 0, stream>>>(xb, wb, ant_b, ht, nullptr, TOK_, MID_, H_, 0);

    // pairwise tanh + BCE + loss
    pair_kernel<<<TOK_, 192, 0, stream>>>(ha, ht, out_w, out_b, target, maskp, output, acc2);
    loss_kernel<<<1, 1, 0, stream>>>(acc2, loss);
}
